// DocumentLevelSelfAttention_31765578121842
// MI455X (gfx1250) — matrix operand order repaired and checked by reading, not compiled, not run on hardware
//
#include <hip/hip_runtime.h>
#include <hip/hip_bf16.h>

// ---------------------------------------------------------------------------
// MI455X (gfx1250): all GEMMs via v_wmma_f32_16x16x32_bf16 (fp32 accumulate),
// HV tiles staged into LDS by the Tensor Data Mover (double-buffered,
// s_wait_tensorcnt) when the TDM builtin is available.
// ---------------------------------------------------------------------------

typedef __attribute__((ext_vector_type(16))) __bf16       v16bf;
typedef __attribute__((ext_vector_type(8)))  float        v8f;
typedef __attribute__((ext_vector_type(4)))  float        v4f;
typedef __attribute__((ext_vector_type(4)))  unsigned int v4u;
typedef __attribute__((ext_vector_type(4)))  unsigned int u32x4;
typedef __attribute__((ext_vector_type(4)))  int          i32x4;
typedef __attribute__((ext_vector_type(8)))  int          i32x8;

constexpr int kB      = 64;             // batch
constexpr int kN      = 1024;           // tokens
constexpr int kD2     = 1024;           // 2*NHID
constexpr int kD3     = 1536;           // 3*NHID
constexpr int kDA     = 350;
constexpr int kDAP    = 384;            // DA padded to 24 WMMA tiles
constexpr int kR      = 30;
constexpr int kMLP    = 2048;
constexpr int kKMLP   = kR * kD3;       // 46080
constexpr int kKSPLIT = 8;
constexpr int kKSEG   = kKMLP / kKSPLIT; // 5760

// TDM tile staging: 32 rows x 32 f32, LDS row stride 128B + 16B pad = 144B
constexpr int kTRow = 144;

#if defined(__gfx1250__) && __has_builtin(__builtin_amdgcn_tensor_load_to_lds) && \
    __has_builtin(__builtin_amdgcn_s_wait_tensorcnt)
#define USE_TDM 1
#else
#define USE_TDM 0
#endif

union Frag {
  v16bf        v;
  unsigned int u[8];
  v4u          q[2];
};

__device__ __forceinline__ unsigned int bf16_rne(float f) {
  unsigned int x = __builtin_bit_cast(unsigned int, f);
  return (x + 0x7fffu + ((x >> 16) & 1u)) >> 16;
}
__device__ __forceinline__ unsigned int pack2(float lo, float hi) {
  return (bf16_rne(lo) & 0xffffu) | (bf16_rne(hi) << 16);
}
__device__ __forceinline__ v8f wmma_bf16(const Frag& a, const Frag& b, v8f c) {
  return __builtin_amdgcn_wmma_f32_16x16x32_bf16(
      /*neg_a=*/false, a.v, /*neg_b=*/false, b.v,
      /*c_mod=*/(short)0, c, /*reuse_a=*/false, /*reuse_b=*/false);
}

// Build a 16x32 bf16 A-fragment from an f32 LDS tile row (ISA 7.12.2 layout:
// lane<16 -> K{0..7,16..23}, lane>=16 -> K{8..15,24..31}).
__device__ __forceinline__ Frag buildA_f32(const unsigned char* rowbase, int hi) {
  Frag a;
  v4f f0 = *(const v4f*)(rowbase + hi * 32);
  v4f f1 = *(const v4f*)(rowbase + hi * 32 + 16);
  v4f f2 = *(const v4f*)(rowbase + 64 + hi * 32);
  v4f f3 = *(const v4f*)(rowbase + 64 + hi * 32 + 16);
  a.u[0] = pack2(f0.x, f0.y); a.u[1] = pack2(f0.z, f0.w);
  a.u[2] = pack2(f1.x, f1.y); a.u[3] = pack2(f1.z, f1.w);
  a.u[4] = pack2(f2.x, f2.y); a.u[5] = pack2(f2.z, f2.w);
  a.u[6] = pack2(f3.x, f3.y); a.u[7] = pack2(f3.z, f3.w);
  return a;
}

#if USE_TDM
// TDM: DMA a 32x32 f32 2D tile (row stride `gstride` elements; 0 = broadcast
// one row 32x) from global into LDS with 16B padding every 128B (bank-safe,
// keeps ds_load_b128 alignment). D# per CDNA5 ISA §8.3/8.4.
__device__ __forceinline__ void tdm_load_tile(const float* gsrc, unsigned gstride,
                                              void* ldst) {
  unsigned long long ga = (unsigned long long)gsrc;
  unsigned ldsa = (unsigned)(unsigned long long)ldst;  // low 32b = LDS offset
  u32x4 g0;
  g0[0] = 1u;                                           // count=1 (valid user D#)
  g0[1] = ldsa;                                         // lds_addr
  g0[2] = (unsigned)(ga & 0xffffffffull);               // global_addr[31:0]
  g0[3] = (unsigned)((ga >> 32) & 0x1ffffffull) | (2u << 30);  // addr[56:32]|type=2
  i32x8 g1;
  g1[0] = (2 << 16) | (1 << 20) | (4 << 22) | (3 << 25); // 4B elem, pad 4dw/32dw
  g1[1] = 0;                                             // tensor_dim0 = 1<<20 (lo16)
  g1[2] = 16;                                            // tensor_dim0 hi, dim1 lo
  g1[3] = 16 | (32 << 16);                               // tensor_dim1 hi, tile_dim0=32
  g1[4] = 32;                                            // tile_dim1=32, tile_dim2=0
  g1[5] = (int)gstride;                                  // tensor_dim0_stride lo
  g1[6] = 0;
  g1[7] = 0;
  i32x4 gz; gz[0] = 0; gz[1] = 0; gz[2] = 0; gz[3] = 0;
#if __clang_major__ >= 23
  i32x8 gz8;
  for (int i = 0; i < 8; ++i) gz8[i] = 0;
  __builtin_amdgcn_tensor_load_to_lds(g0, g1, gz, gz, gz8, 0);
#else
  __builtin_amdgcn_tensor_load_to_lds(g0, g1, gz, gz, 0);
#endif
}
#endif

// ---------------------------------------------------------------------------
// Kernel 0: weights -> bf16, zero-padded (W1:[384,1536], W2:[32,384])
// ---------------------------------------------------------------------------
__global__ __launch_bounds__(256) void pack_weights_kernel(
    const float* __restrict__ W1, const float* __restrict__ W2,
    unsigned short* __restrict__ W1bf, unsigned short* __restrict__ W2bf) {
  int i = blockIdx.x * 256 + threadIdx.x;
  constexpr int n1 = kDAP * kD3;
  if (i < n1) {
    int a = i / kD3, d = i - a * kD3;
    float v = (a < kDA) ? W1[(size_t)a * kD3 + d] : 0.0f;
    W1bf[i] = (unsigned short)bf16_rne(v);
  } else if (i < n1 + 32 * kDAP) {
    int j = i - n1;
    int r = j / kDAP, a = j - r * kDAP;
    float v = (r < kR && a < kDA) ? W2[(size_t)r * kDA + a] : 0.0f;
    W2bf[j] = (unsigned short)bf16_rne(v);
  }
}

// ---------------------------------------------------------------------------
// Kernel 1: fused s2 = W2 @ tanh(W1 @ HV^T), scores stored [B][R][N].
// Grid (N/32, B), 8 waves. 32 n-rows per block: one B-fragment of W1 feeds
// two WMMAs (halves L2 traffic on W1). HV f32 tiles arrive via TDM
// (double-buffered, s_wait_tensorcnt) or a cooperative-fill fallback.
// ---------------------------------------------------------------------------
__global__ __launch_bounds__(256) void scores_kernel(
    const float* __restrict__ enc, const float* __restrict__ cat,
    const unsigned short* __restrict__ W1bf,
    const unsigned short* __restrict__ W2bf,
    float* __restrict__ scoresT) {
  const int b    = blockIdx.y;
  const int n0   = blockIdx.x * 32;
  const int tid  = threadIdx.x;
  const int lane = tid & 31;
  const int wave = tid >> 5;
  const int row  = lane & 15;
  const int hi   = lane >> 4;

  __shared__ __align__(16) unsigned char sT[2][32 * kTRow];  // f32 HV tiles
  __shared__ __align__(16) unsigned char sS1[32 * 784];      // 32 x 384 bf16

  const float* encb = enc + (size_t)b * kN * kD2;
  const float* catb = cat + (size_t)b * 512;
  const int colbase = wave * 48;

  v8f acc[2][3] = {};

#if USE_TDM
  if (wave == 0)
    tdm_load_tile(encb + (size_t)n0 * kD2, kD2, &sT[0][0]);  // kc = 0 tile
  int p = 0;
#endif

  for (int kc = 0; kc < kD3; kc += 32) {
#if USE_TDM
    if (wave == 0) __builtin_amdgcn_s_wait_tensorcnt(0);
    __syncthreads();                       // tile[p] ready for everyone
    if (wave == 0 && kc + 32 < kD3) {      // prefetch next tile into tile[p^1]
      int kn = kc + 32;
      const float* gsrc = (kn < kD2) ? (encb + (size_t)n0 * kD2 + kn)
                                     : (catb + (kn - kD2));
      tdm_load_tile(gsrc, (kn < kD2) ? kD2 : 0u, &sT[p ^ 1][0]);
    }
    const unsigned char* tb = &sT[p][0];
#else
    __syncthreads();                       // previous reads done
    {                                      // cooperative fill: 1024 f32 / 256 thr
      int e = tid * 4, r = e >> 5, c = e & 31;
      int k = kc + c;
      v4f f;
      if (k < kD2) f = *(const v4f*)(encb + (size_t)(n0 + r) * kD2 + k);
      else         f = *(const v4f*)(catb + (k - kD2));
      *(v4f*)(&sT[0][0] + r * kTRow + c * 4) = f;
    }
    __syncthreads();
    const unsigned char* tb = &sT[0][0];
#endif

    Frag a0 = buildA_f32(tb + (size_t)row * kTRow, hi);          // n-rows 0..15
    Frag a1 = buildA_f32(tb + (size_t)(16 + row) * kTRow, hi);   // n-rows 16..31
#pragma unroll
    for (int t = 0; t < 3; ++t) {
      Frag bw;  // B 32x16 of W1: lane col = row, K = kc + {hi*8+0..7, 16+hi*8+0..7}
      const unsigned short* wp =
          W1bf + (size_t)(colbase + t * 16 + row) * kD3 + kc + hi * 8;
      bw.q[0] = *(const v4u*)(wp);
      bw.q[1] = *(const v4u*)(wp + 16);
      acc[0][t] = wmma_bf16(a0, bw, acc[0][t]);
      acc[1][t] = wmma_bf16(a1, bw, acc[1][t]);
    }
#if USE_TDM
    p ^= 1;
#endif
  }

  // tanh, C-tiles (m = j + hi*8, col = lane&15) -> LDS bf16
#pragma unroll
  for (int nt = 0; nt < 2; ++nt) {
#pragma unroll
    for (int t = 0; t < 3; ++t) {
#pragma unroll
      for (int j = 0; j < 8; ++j) {
        int m    = nt * 16 + j + hi * 8;
        int colc = colbase + t * 16 + row;
        *(unsigned short*)(sS1 + m * 784 + colc * 2) =
            (unsigned short)bf16_rne(tanhf(acc[nt][t][j]));
      }
    }
  }
  __syncthreads();

  if (wave < 4) {  // s2: (n-tile, r-tile) per wave, K = 384
    const int nt = wave >> 1;
    const int rc = (wave & 1) * 16;
    v8f cs = {};
    for (int ka = 0; ka < kDAP; ka += 32) {
      Frag a2;
      const unsigned char* pA = sS1 + (nt * 16 + row) * 784 + (ka + hi * 8) * 2;
      a2.q[0] = *(const v4u*)(pA);
      a2.q[1] = *(const v4u*)(pA + 32);
      Frag b2;
      const unsigned short* wp = W2bf + (size_t)(rc + row) * kDAP + ka + hi * 8;
      b2.q[0] = *(const v4u*)(wp);
      b2.q[1] = *(const v4u*)(wp + 16);
      cs = wmma_bf16(a2, b2, cs);
    }
    const int r = rc + row;
    if (r < kR) {
#pragma unroll
      for (int j = 0; j < 8; ++j) {
        int m = nt * 16 + j + hi * 8;
        scoresT[((size_t)b * kR + r) * kN + n0 + m] = cs[j];
      }
    }
  }
}

// ---------------------------------------------------------------------------
// Kernel 2: length-masked softmax over N per (b,r). Writes A into d_out.
// ---------------------------------------------------------------------------
__device__ __forceinline__ float waveMax(float v) {
  for (int o = 16; o > 0; o >>= 1) v = fmaxf(v, __shfl_xor(v, o, 32));
  return v;
}
__device__ __forceinline__ float waveSum(float v) {
  for (int o = 16; o > 0; o >>= 1) v += __shfl_xor(v, o, 32);
  return v;
}

__global__ __launch_bounds__(256) void softmax_kernel(
    const float* __restrict__ scoresT, const int* __restrict__ lens,
    float* __restrict__ Aout) {
  const int r = blockIdx.x, b = blockIdx.y;
  const int len = lens[b];
  const float* srow = scoresT + ((size_t)b * kR + r) * kN;
  float* arow = Aout + ((size_t)b * kR + r) * kN;
  const int tid = threadIdx.x;

  v4f v = *(const v4f*)(srow + tid * 4);
  float x[4] = {v.x, v.y, v.z, v.w};
  float mx = -3.0e38f;
#pragma unroll
  for (int i = 0; i < 4; ++i) {
    if (tid * 4 + i >= len) x[i] = -1e9f;  // NEG from reference
    mx = fmaxf(mx, x[i]);
  }
  mx = waveMax(mx);

  __shared__ float red[8];
  __shared__ float bcast;
  if ((tid & 31) == 0) red[tid >> 5] = mx;
  __syncthreads();
  if (tid == 0) {
    float m = red[0];
#pragma unroll
    for (int i = 1; i < 8; ++i) m = fmaxf(m, red[i]);
    bcast = m;
  }
  __syncthreads();
  mx = bcast;

  float e[4], s = 0.0f;
#pragma unroll
  for (int i = 0; i < 4; ++i) { e[i] = __expf(x[i] - mx); s += e[i]; }
  s = waveSum(s);
  if ((tid & 31) == 0) red[tid >> 5] = s;
  __syncthreads();
  if (tid == 0) {
    float t = 0.0f;
#pragma unroll
    for (int i = 0; i < 8; ++i) t += red[i];
    bcast = t;
  }
  __syncthreads();
  const float inv = 1.0f / bcast;

  v4f o; o.x = e[0] * inv; o.y = e[1] * inv; o.z = e[2] * inv; o.w = e[3] * inv;
  *(v4f*)(arow + tid * 4) = o;
}

// ---------------------------------------------------------------------------
// Kernel 3: M = A @ HV  ([B,30,1536], K = N = 1024), result bf16 to ws.
// Grid (24, B): block covers 32 r-rows x 64 d-cols; reads enc exactly once.
// ---------------------------------------------------------------------------
__global__ __launch_bounds__(256) void attn_apply_kernel(
    const float* __restrict__ enc, const float* __restrict__ cat,
    const float* __restrict__ A, unsigned short* __restrict__ Mbf) {
  const int b    = blockIdx.y;
  const int slab = blockIdx.x;
  const int tid  = threadIdx.x;
  const int lane = tid & 31;
  const int wave = tid >> 5;
  const int row  = lane & 15;
  const int hi   = lane >> 4;
  const int rt   = wave & 1;
  const int cb   = slab * 64 + (wave >> 1) * 16;

  const int  arow_idx = rt * 16 + row;
  const bool avalid   = arow_idx < kR;
  const float* ap = A + ((size_t)b * kR + (avalid ? arow_idx : 0)) * kN;
  const int d = cb + row;
  const float* encb = enc + (size_t)b * kN * kD2 + d;
  const bool isenc = d < kD2;
  const float catval = isenc ? 0.0f : cat[(size_t)b * 512 + (d - kD2)];

  v8f c = {};
  for (int kc = 0; kc < kN; kc += 32) {
    Frag a;
    if (avalid) {
      const float* p = ap + kc + hi * 16;
      v4f f0 = *(const v4f*)(p);
      v4f f1 = *(const v4f*)(p + 4);
      v4f f2 = *(const v4f*)(p + 8);
      v4f f3 = *(const v4f*)(p + 12);
      a.u[0] = pack2(f0.x, f0.y); a.u[1] = pack2(f0.z, f0.w);
      a.u[2] = pack2(f1.x, f1.y); a.u[3] = pack2(f1.z, f1.w);
      a.u[4] = pack2(f2.x, f2.y); a.u[5] = pack2(f2.z, f2.w);
      a.u[6] = pack2(f3.x, f3.y); a.u[7] = pack2(f3.z, f3.w);
    } else {
#pragma unroll
      for (int j = 0; j < 8; ++j) a.u[j] = 0u;
    }
    Frag bw;
    const int kb0 = kc + hi * 16;
    if (isenc) {
#pragma unroll
      for (int j = 0; j < 8; ++j) {  // lanes -> consecutive d: coalesced
        float g0 = encb[(size_t)(kb0 + 2 * j) * kD2];
        float g1 = encb[(size_t)(kb0 + 2 * j + 1) * kD2];
        bw.u[j] = pack2(g0, g1);
      }
    } else {
      unsigned int pk = pack2(catval, catval);
#pragma unroll
      for (int j = 0; j < 8; ++j) bw.u[j] = pk;
    }
    c = wmma_bf16(a, bw, c);
  }
#pragma unroll
  for (int j = 0; j < 8; ++j) {
    int rout = rt * 16 + j + hi * 8;
    if (rout < kR)
      Mbf[(size_t)b * kKMLP + (size_t)rout * kD3 + cb + row] =
          (unsigned short)bf16_rne(c[j]);
  }
}

// ---------------------------------------------------------------------------
// Kernel 4: out = M @ W_mlp^T + b. K = 46080, split 8 ways (deterministic
// partial buffer, no float atomics). W_mlp streamed once; prefetch ahead.
// ---------------------------------------------------------------------------
__global__ __launch_bounds__(256) void mlp_kernel(
    const unsigned short* __restrict__ Mbf, const float* __restrict__ Wm,
    float* __restrict__ partial) {
  const int slab = blockIdx.x;   // 64 slabs of 32 output cols
  const int ks   = blockIdx.y;   // K split
  const int tid  = threadIdx.x;
  const int lane = tid & 31;
  const int wave = tid >> 5;
  const int row  = lane & 15;
  const int hi   = lane >> 4;
  const int bt   = wave & 3;
  const int ct   = slab * 32 + (wave >> 2) * 16;

  const int brow = bt * 16 + row;
  const int ocol = ct + row;
  const unsigned short* mrow = Mbf + (size_t)brow * kKMLP;
  const float* wrow = Wm + (size_t)ocol * kKMLP;
  const int k0 = ks * kKSEG, k1 = k0 + kKSEG;

  v8f c = {};
  for (int kc = k0; kc < k1; kc += 32) {
    __builtin_prefetch(wrow + kc + 1024, 0, 1);  // -> global_prefetch_b8
    Frag a;
    {
      const unsigned short* p = mrow + kc + hi * 16;
      a.q[0] = *(const v4u*)(p);
      a.q[1] = *(const v4u*)(p + 8);
    }
    Frag bw;
    {
      const float* p = wrow + kc + hi * 16;
      v4f f0 = *(const v4f*)(p);
      v4f f1 = *(const v4f*)(p + 4);
      v4f f2 = *(const v4f*)(p + 8);
      v4f f3 = *(const v4f*)(p + 12);
      bw.u[0] = pack2(f0.x, f0.y); bw.u[1] = pack2(f0.z, f0.w);
      bw.u[2] = pack2(f1.x, f1.y); bw.u[3] = pack2(f1.z, f1.w);
      bw.u[4] = pack2(f2.x, f2.y); bw.u[5] = pack2(f2.z, f2.w);
      bw.u[6] = pack2(f3.x, f3.y); bw.u[7] = pack2(f3.z, f3.w);
    }
    c = wmma_bf16(a, bw, c);
  }
#pragma unroll
  for (int j = 0; j < 8; ++j) {
    int bb = bt * 16 + j + hi * 8;
    partial[((size_t)ks * kB + bb) * kMLP + ct + row] = c[j];
  }
}

__global__ __launch_bounds__(256) void mlp_reduce_kernel(
    const float* __restrict__ partial, const float* __restrict__ bias,
    float* __restrict__ out) {
  int i = blockIdx.x * 256 + threadIdx.x;  // kB*kMLP
  float s = bias[i & (kMLP - 1)];
#pragma unroll
  for (int ks = 0; ks < kKSPLIT; ++ks) s += partial[(size_t)ks * kB * kMLP + i];
  out[i] = s;
}

// ---------------------------------------------------------------------------
extern "C" void kernel_launch(void* const* d_in, const int* in_sizes, int n_in,
                              void* d_out, int out_size, void* d_ws, size_t ws_size,
                              hipStream_t stream) {
  (void)in_sizes; (void)n_in; (void)out_size; (void)ws_size;
  const float* enc  = (const float*)d_in[0];
  const int*   lens = (const int*)d_in[1];
  // d_in[2] = batch_size scalar (compile-time constant here)
  const float* cat  = (const float*)d_in[3];
  const float* W1   = (const float*)d_in[4];
  const float* W2   = (const float*)d_in[5];
  const float* Wm   = (const float*)d_in[6];
  const float* bm   = (const float*)d_in[7];

  // workspace layout (~19 MB total, 16B-aligned sections)
  unsigned short* W1bf    = (unsigned short*)d_ws;                // 384*1536
  unsigned short* W2bf    = W1bf + (size_t)kDAP * kD3;            // 32*384
  float*          scoresT = (float*)(W2bf + 32 * kDAP);           // B*R*N
  unsigned short* Mbf     = (unsigned short*)(scoresT + (size_t)kB * kR * kN);
  float*          partial = (float*)(Mbf + (size_t)kB * kKMLP);   // 8*B*MLP

  float* out  = (float*)d_out;
  float* Aout = out + (size_t)kB * kMLP;  // A follows `out` in d_out

  pack_weights_kernel<<<(kDAP * kD3 + 32 * kDAP + 255) / 256, 256, 0, stream>>>(
      W1, W2, W1bf, W2bf);
  scores_kernel<<<dim3(kN / 32, kB), 256, 0, stream>>>(enc, cat, W1bf, W2bf, scoresT);
  softmax_kernel<<<dim3(kR, kB), 256, 0, stream>>>(scoresT, lens, Aout);
  attn_apply_kernel<<<dim3(kD3 / 64, kB), 256, 0, stream>>>(enc, cat, Aout, Mbf);
  mlp_kernel<<<dim3(kMLP / 32, kKSPLIT), 256, 0, stream>>>(Mbf, Wm, partial);
  mlp_reduce_kernel<<<(kB * kMLP) / 256, 256, 0, stream>>>(partial, bm, out);
}
